// GGNN_FCModel_79001628442641
// MI455X (gfx1250) — matrix operand order repaired and checked by reading, not compiled, not run on hardware
//
#include <hip/hip_runtime.h>
#include <hip/hip_bf16.h>

// ---------------- problem constants (match reference) ----------------
#define N_NODES   50000
#define E_EDGES   800000
#define HDIM      256
#define T_TYPES   4
#define L_LAYERS  2
#define S_STEPS   3
#define C_CLASSES 3
#define ROW_TILES   (N_NODES / 16)          // 3125 (16-row tiles, exact)
#define ROW_PAIRS   ((ROW_TILES + 1) / 2)   // 1563 (32-row tiles, last is half)

typedef __bf16 bf16_t;
typedef __attribute__((ext_vector_type(16))) __bf16 v16bf;
typedef __attribute__((ext_vector_type(8)))  __bf16 v8bf;
typedef __attribute__((ext_vector_type(8)))  float  v8f;
typedef __attribute__((ext_vector_type(4)))  unsigned int u32x4;

union V16 { u32x4 u[2]; v16bf v; };
union V8  { u32x4 u;    v8bf  v; };

#define WMMA_BF16(A, B, C) \
  __builtin_amdgcn_wmma_f32_16x16x32_bf16(false, (A), false, (B), (short)0, (C), false, false)

// A fragment (16x32, row-major source, leading dim = ldk elements):
// lane L holds row M = L&15; lanes 0-15: K 0-7 and 16-23; lanes 16-31: K 8-15 and 24-31.
__device__ __forceinline__ v16bf load_a_frag(const bf16_t* __restrict__ base, int ldk, int lane) {
  const int m  = lane & 15;
  const int kb = (lane < 16) ? 0 : 8;
  V16 r;
  r.u[0] = *(const u32x4*)(base + (size_t)m * ldk + kb);
  r.u[1] = *(const u32x4*)(base + (size_t)m * ldk + kb + 16);
  return r.v;
}

// B fragment (32x16). B[k][n] = W[n0+n][k0+k] for y = x @ W.T with W row-major [out, in]:
// lane L holds column N = L&15; lanes 0-15: K 0-7 and 16-23; lanes 16-31: K 8-15 and 24-31.
__device__ __forceinline__ v16bf load_b_frag(const bf16_t* __restrict__ base, int ldk, int lane) {
  const int n  = lane & 15;
  const int kb = (lane < 16) ? 0 : 8;
  const bf16_t* p = base + (size_t)n * ldk + kb;
  V16 r;
  r.u[0] = *(const u32x4*)(p);
  r.u[1] = *(const u32x4*)(p + 16);
  return r.v;
}

// ---------------- elementwise helpers ----------------
__global__ __launch_bounds__(256) void cvt_f32_bf16(const float* __restrict__ s,
                                                    bf16_t* __restrict__ d, long long n) {
  long long i = (long long)blockIdx.x * blockDim.x + threadIdx.x;
  long long stride = (long long)gridDim.x * blockDim.x;
  for (; i < n; i += stride) d[i] = (bf16_t)s[i];
}

// ---------------- per-etype transform: Wh[t] = h @ W[t].T + b[t], stored bf16 ----------------
// one wave per 32(M)x64(N) output tile; 8 f32 accumulators; B fragments amortized over 2 M-subtiles.
__global__ __launch_bounds__(256) void etype_gemm(const bf16_t* __restrict__ hb,   // [N,H] bf16
                                                  const bf16_t* __restrict__ Wb,   // [T,H,H] bf16
                                                  const float*  __restrict__ bias, // [T,H]
                                                  bf16_t* __restrict__ Whb)        // [T,N,H] bf16
{
  const int lane = threadIdx.x & 31;
  const int wid  = blockIdx.x * 8 + (threadIdx.x >> 5);
  if (wid >= T_TYPES * ROW_PAIRS * 4) return;
  const int t   = wid / (ROW_PAIRS * 4);
  const int rem = wid - t * (ROW_PAIRS * 4);
  const int m0  = (rem >> 2) * 32;
  const int n0  = (rem & 3) * 64;
  const bool v1 = (m0 + 16) < N_NODES;        // second M-subtile valid?
  const int  m1 = v1 ? (m0 + 16) : m0;        // clamp loads for the tail tile

  const bf16_t* Wt = Wb + (size_t)t * HDIM * HDIM;
  v8f acc0[4] = {}, acc1[4] = {};
#pragma unroll
  for (int k0 = 0; k0 < HDIM; k0 += 32) {
    v16bf a0 = load_a_frag(hb + (size_t)m0 * HDIM + k0, HDIM, lane);
    v16bf a1 = load_a_frag(hb + (size_t)m1 * HDIM + k0, HDIM, lane);
#pragma unroll
    for (int j = 0; j < 4; ++j) {
      v16bf b = load_b_frag(Wt + (size_t)(n0 + j * 16) * HDIM + k0, HDIM, lane);
      acc0[j] = WMMA_BF16(a0, b, acc0[j]);
      acc1[j] = WMMA_BF16(a1, b, acc1[j]);
    }
  }

  const int mofs = (lane < 16) ? 0 : 8;
  const int col  = lane & 15;
  const bf16_t* __restrict__ dummy = nullptr; (void)dummy;
#pragma unroll
  for (int j = 0; j < 4; ++j) {
    const int c = n0 + j * 16 + col;
    const float bv = bias[t * HDIM + c];
    bf16_t* outc0 = Whb + (size_t)t * N_NODES * HDIM + (size_t)m0 * HDIM + c;
#pragma unroll
    for (int r = 0; r < 8; ++r)
      outc0[(size_t)(r + mofs) * HDIM] = (bf16_t)(acc0[j][r] + bv);
    if (v1) {
      bf16_t* outc1 = Whb + (size_t)t * N_NODES * HDIM + (size_t)m1 * HDIM + c;
#pragma unroll
      for (int r = 0; r < 8; ++r)
        outc1[(size_t)(r + mofs) * HDIM] = (bf16_t)(acc1[j][r] + bv);
    }
  }
}

// ---------------- edge gather + scatter-add (f32 atomics into a[N,H]) ----------------
// one wave per edge; lane handles 8 columns (16B bf16 load -> 8 atomic f32 adds, L2-local).
__global__ __launch_bounds__(256) void scatter_edges(const int* __restrict__ src,
                                                     const int* __restrict__ dst,
                                                     const int* __restrict__ et,
                                                     const bf16_t* __restrict__ Whb,
                                                     float* __restrict__ a)
{
  const int wid  = (int)(((long long)blockIdx.x * blockDim.x + threadIdx.x) >> 5);
  const int lane = threadIdx.x & 31;
  if (wid >= E_EDGES) return;
  const int s = src[wid], d = dst[wid], t = et[wid];

  V8 m;
  m.u = *(const u32x4*)(Whb + ((size_t)t * N_NODES + s) * HDIM + lane * 8);
  float* arow = a + (size_t)d * HDIM + lane * 8;
#pragma unroll
  for (int i = 0; i < 8; ++i) atomicAdd(arow + i, (float)m.v[i]);
}

// ---------------- fused GRU epilogue for one 16x16 sub-tile ----------------
__device__ __forceinline__ void gru_epilogue(int mbase, int col, int mofs,
                                             const v8f& xr, const v8f& xz, const v8f& xn,
                                             const v8f& hr, const v8f& hz, const v8f& hn,
                                             float bir, float biz, float bin,
                                             float bhr, float bhz, float bhn,
                                             float* __restrict__ h,
                                             bf16_t* __restrict__ hb_out)
{
#pragma unroll
  for (int r = 0; r < 8; ++r) {
    const size_t idx = (size_t)(mbase + mofs + r) * HDIM + col;
    const float rg = 1.f / (1.f + __expf(-(xr[r] + bir + hr[r] + bhr)));
    const float zg = 1.f / (1.f + __expf(-(xz[r] + biz + hz[r] + bhz)));
    const float ng = tanhf(xn[r] + bin + rg * (hn[r] + bhn));
    const float hv = h[idx];
    const float hnew = (1.f - zg) * ng + zg * hv;
    h[idx]      = hnew;
    hb_out[idx] = (bf16_t)hnew;
  }
}

// ---------------- fused GRU cell ----------------
// one wave per 32(node)x16(col) tile: twelve WMMA chains (r/z/n x input/hidden x 2 M-subtiles,
// K=256); weight fragments are loaded once per k-step and reused for both M-subtiles.
__global__ __launch_bounds__(256) void gru_fused(const bf16_t* __restrict__ ab,    // [N,H] bf16 (aggregated messages)
                                                 const bf16_t* __restrict__ hb,    // [N,H] bf16 (current state)
                                                 float*        __restrict__ h,     // [N,H] f32 state, updated in place
                                                 const bf16_t* __restrict__ wih,   // [3H,H] bf16
                                                 const bf16_t* __restrict__ whh,   // [3H,H] bf16
                                                 const float*  __restrict__ bih,   // [3H]
                                                 const float*  __restrict__ bhh,   // [3H]
                                                 bf16_t*       __restrict__ hb_out)// [N,H] bf16 next state
{
  const int lane = threadIdx.x & 31;
  const int wid  = blockIdx.x * 8 + (threadIdx.x >> 5);
  if (wid >= ROW_PAIRS * 16) return;
  const int m0 = (wid >> 4) * 32;
  const int j0 = (wid & 15) * 16;
  const bool v1 = (m0 + 16) < N_NODES;
  const int  m1 = v1 ? (m0 + 16) : m0;

  v8f xr0 = {}, xz0 = {}, xn0 = {}, hr0 = {}, hz0 = {}, hn0 = {};
  v8f xr1 = {}, xz1 = {}, xn1 = {}, hr1 = {}, hz1 = {}, hn1 = {};
#pragma unroll
  for (int k0 = 0; k0 < HDIM; k0 += 32) {
    v16bf Aa0 = load_a_frag(ab + (size_t)m0 * HDIM + k0, HDIM, lane);
    v16bf Ah0 = load_a_frag(hb + (size_t)m0 * HDIM + k0, HDIM, lane);
    v16bf Aa1 = load_a_frag(ab + (size_t)m1 * HDIM + k0, HDIM, lane);
    v16bf Ah1 = load_a_frag(hb + (size_t)m1 * HDIM + k0, HDIM, lane);

    v16bf Br = load_b_frag(wih + (size_t)(0 * HDIM + j0) * HDIM + k0, HDIM, lane);
    v16bf Bz = load_b_frag(wih + (size_t)(1 * HDIM + j0) * HDIM + k0, HDIM, lane);
    v16bf Bn = load_b_frag(wih + (size_t)(2 * HDIM + j0) * HDIM + k0, HDIM, lane);
    xr0 = WMMA_BF16(Aa0, Br, xr0);  xr1 = WMMA_BF16(Aa1, Br, xr1);
    xz0 = WMMA_BF16(Aa0, Bz, xz0);  xz1 = WMMA_BF16(Aa1, Bz, xz1);
    xn0 = WMMA_BF16(Aa0, Bn, xn0);  xn1 = WMMA_BF16(Aa1, Bn, xn1);

    v16bf Cr = load_b_frag(whh + (size_t)(0 * HDIM + j0) * HDIM + k0, HDIM, lane);
    v16bf Cz = load_b_frag(whh + (size_t)(1 * HDIM + j0) * HDIM + k0, HDIM, lane);
    v16bf Cn = load_b_frag(whh + (size_t)(2 * HDIM + j0) * HDIM + k0, HDIM, lane);
    hr0 = WMMA_BF16(Ah0, Cr, hr0);  hr1 = WMMA_BF16(Ah1, Cr, hr1);
    hz0 = WMMA_BF16(Ah0, Cz, hz0);  hz1 = WMMA_BF16(Ah1, Cz, hz1);
    hn0 = WMMA_BF16(Ah0, Cn, hn0);  hn1 = WMMA_BF16(Ah1, Cn, hn1);
  }

  const int mofs = (lane < 16) ? 0 : 8;
  const int col  = j0 + (lane & 15);
  const float bir = bih[col], biz = bih[HDIM + col], bin = bih[2 * HDIM + col];
  const float bhr = bhh[col], bhz = bhh[HDIM + col], bhn = bhh[2 * HDIM + col];

  gru_epilogue(m0, col, mofs, xr0, xz0, xn0, hr0, hz0, hn0,
               bir, biz, bin, bhr, bhz, bhn, h, hb_out);
  if (v1)
    gru_epilogue(m1, col, mofs, xr1, xz1, xn1, hr1, hz1, hn1,
                 bir, biz, bin, bhr, bhz, bhn, h, hb_out);
}

// ---------------- mean pool over nodes (column sums via f32 atomics) ----------------
__global__ __launch_bounds__(256) void pool_cols(const float* __restrict__ h,
                                                 float* __restrict__ pooled) {
  const int t = threadIdx.x;           // column 0..255
  float s = 0.f;
  for (int n = blockIdx.x; n < N_NODES; n += gridDim.x) s += h[(size_t)n * HDIM + t];
  atomicAdd(pooled + t, s);
}

__global__ void final_fc(const float* __restrict__ pooled, const float* __restrict__ fw,
                         const float* __restrict__ fb, float* __restrict__ out) {
  const int c = threadIdx.x;
  if (c >= C_CLASSES) return;
  float s = 0.f;
  for (int k = 0; k < HDIM; ++k) s += pooled[k] * fw[c * HDIM + k];
  out[c] = fb[c] + s * (1.0f / (float)N_NODES);
}

// ---------------- host side ----------------
static inline size_t align_up(size_t x) { return (x + 255) & ~(size_t)255; }

extern "C" void kernel_launch(void* const* d_in, const int* in_sizes, int n_in,
                              void* d_out, int out_size, void* d_ws, size_t ws_size,
                              hipStream_t stream) {
  (void)in_sizes; (void)n_in; (void)out_size; (void)ws_size;

  const float* x     = (const float*)d_in[0];
  const int*   src   = (const int*)d_in[1];
  const int*   dst   = (const int*)d_in[2];
  const int*   etype = (const int*)d_in[3];
  const float* W_lin = (const float*)d_in[4];   // [L,T,H,H]
  const float* b_lin = (const float*)d_in[5];   // [L,T,H]
  const float* w_ih  = (const float*)d_in[6];   // [L,3H,H]
  const float* w_hh  = (const float*)d_in[7];   // [L,3H,H]
  const float* b_ih  = (const float*)d_in[8];   // [L,3H]
  const float* b_hh  = (const float*)d_in[9];   // [L,3H]
  const float* fc_w  = (const float*)d_in[10];  // [C,H]
  const float* fc_b  = (const float*)d_in[11];  // [C]
  float* out = (float*)d_out;

  const size_t NH = (size_t)N_NODES * HDIM;

  // ---- carve workspace ----
  char* ws = (char*)d_ws;
  size_t off = 0;
  auto carve = [&](size_t bytes) -> void* { void* p = ws + off; off = align_up(off + bytes); return p; };

  float*  h_f32   = (float*)carve(NH * sizeof(float));
  bf16_t* hb[2]   = { (bf16_t*)carve(NH * sizeof(bf16_t)),
                      (bf16_t*)carve(NH * sizeof(bf16_t)) };
  float*  a_f32   = (float*)carve(NH * sizeof(float));
  bf16_t* a_bf    = (bf16_t*)carve(NH * sizeof(bf16_t));
  bf16_t* Whb     = (bf16_t*)carve((size_t)T_TYPES * NH * sizeof(bf16_t));
  bf16_t* Wlin_bf = (bf16_t*)carve((size_t)L_LAYERS * T_TYPES * HDIM * HDIM * sizeof(bf16_t));
  bf16_t* wih_bf  = (bf16_t*)carve((size_t)L_LAYERS * 3 * HDIM * HDIM * sizeof(bf16_t));
  bf16_t* whh_bf  = (bf16_t*)carve((size_t)L_LAYERS * 3 * HDIM * HDIM * sizeof(bf16_t));
  float*  pooled  = (float*)carve(HDIM * sizeof(float));

  // ---- one-time per launch: f32 state copy + bf16 weight conversion ----
  hipMemcpyAsync(h_f32, x, NH * sizeof(float), hipMemcpyDeviceToDevice, stream);
  cvt_f32_bf16<<<1024, 256, 0, stream>>>(W_lin, Wlin_bf, (long long)L_LAYERS * T_TYPES * HDIM * HDIM);
  cvt_f32_bf16<<<1024, 256, 0, stream>>>(w_ih,  wih_bf,  (long long)L_LAYERS * 3 * HDIM * HDIM);
  cvt_f32_bf16<<<1024, 256, 0, stream>>>(w_hh,  whh_bf,  (long long)L_LAYERS * 3 * HDIM * HDIM);
  cvt_f32_bf16<<<2048, 256, 0, stream>>>(x, hb[0], (long long)NH);

  const int gemm_blocks    = (T_TYPES * ROW_PAIRS * 4 + 7) / 8;   // 8 waves/block
  const int gru_blocks     = (ROW_PAIRS * 16 + 7) / 8;
  const int scatter_blocks = (E_EDGES * 32 + 255) / 256;

  int cur = 0;
  for (int l = 0; l < L_LAYERS; ++l) {
    const bf16_t* Wl  = Wlin_bf + (size_t)l * T_TYPES * HDIM * HDIM;
    const float*  bl  = b_lin   + (size_t)l * T_TYPES * HDIM;
    const bf16_t* wil = wih_bf  + (size_t)l * 3 * HDIM * HDIM;
    const bf16_t* whl = whh_bf  + (size_t)l * 3 * HDIM * HDIM;
    const float*  bil = b_ih    + (size_t)l * 3 * HDIM;
    const float*  bhl = b_hh    + (size_t)l * 3 * HDIM;

    for (int s = 0; s < S_STEPS; ++s) {
      etype_gemm<<<gemm_blocks, 256, 0, stream>>>(hb[cur], Wl, bl, Whb);
      hipMemsetAsync(a_f32, 0, NH * sizeof(float), stream);
      scatter_edges<<<scatter_blocks, 256, 0, stream>>>(src, dst, etype, Whb, a_f32);
      cvt_f32_bf16<<<2048, 256, 0, stream>>>(a_f32, a_bf, (long long)NH);
      gru_fused<<<gru_blocks, 256, 0, stream>>>(a_bf, hb[cur], h_f32,
                                                wil, whl, bil, bhl, hb[cur ^ 1]);
      cur ^= 1;
    }
  }

  hipMemsetAsync(pooled, 0, HDIM * sizeof(float), stream);
  pool_cols<<<512, 256, 0, stream>>>(h_f32, pooled);
  final_fc<<<1, 32, 0, stream>>>(pooled, fc_w, fc_b, out);
}
